// SoftQuantizeEMAReset_68023692034354
// MI455X (gfx1250) — matrix operand order repaired and checked by reading, not compiled, not run on hardware
//
#include <hip/hip_runtime.h>

// ---------------- problem dims ----------------
#define NN 64
#define CC 512
#define TT 1024
#define NT (NN * TT)   // 65536 rows
#define AD 512         // attn dim
#define KC 1024        // codebook entries
#define INV_NT (1.0f / 65536.0f)

// ---------------- types ----------------
typedef __bf16 v16bf __attribute__((ext_vector_type(16)));
typedef __bf16 v8bf  __attribute__((ext_vector_type(8)));
typedef float  v8f   __attribute__((ext_vector_type(8)));
typedef int    v4i_t __attribute__((ext_vector_type(4)));

#if defined(__HIP_DEVICE_COMPILE__) && __has_builtin(__builtin_amdgcn_global_load_async_to_lds_b128)
#define USE_ASYNC 1
#else
#define USE_ASYNC 0
#endif

__device__ __forceinline__ unsigned short f2bf(float f) {
    unsigned int u = __float_as_uint(f);
    u += 0x7FFFu + ((u >> 16) & 1u);   // round-to-nearest-even
    return (unsigned short)(u >> 16);
}

#if USE_ASYNC
typedef __attribute__((address_space(1))) v4i_t* g_v4i_ptr;
typedef __attribute__((address_space(3))) v4i_t* l_v4i_ptr;

__device__ __forceinline__ void async_copy_b128(const unsigned short* g, unsigned short* l) {
    unsigned short* gnc = const_cast<unsigned short*>(g);
    __builtin_amdgcn_global_load_async_to_lds_b128(
        (g_v4i_ptr)gnc, (l_v4i_ptr)l, 0, 0);
}
__device__ __forceinline__ void wait_async() {
#if __has_builtin(__builtin_amdgcn_s_wait_asynccnt)
    __builtin_amdgcn_s_wait_asynccnt(0);
#else
    asm volatile("s_wait_asynccnt 0x0" ::: "memory");
#endif
}
#endif

// ---------------- block reductions (wave32) ----------------
__device__ __forceinline__ float block_reduce_sum(float v) {
    __shared__ float sbuf[9];
    #pragma unroll
    for (int off = 16; off; off >>= 1) v += __shfl_xor(v, off, 32);
    __syncthreads();
    if ((threadIdx.x & 31) == 0) sbuf[threadIdx.x >> 5] = v;
    __syncthreads();
    if (threadIdx.x == 0) {
        float s = 0.f;
        int nw = (blockDim.x + 31) >> 5;
        for (int w = 0; w < nw; ++w) s += sbuf[w];
        sbuf[8] = s;
    }
    __syncthreads();
    float r = sbuf[8];
    __syncthreads();
    return r;
}

__device__ __forceinline__ void block_argmax(float v, int idx, float& bv_out, int& bi_out) {
    __shared__ float sv[8];
    __shared__ int   si[8];
    __shared__ float bv;
    __shared__ int   bi;
    #pragma unroll
    for (int off = 16; off; off >>= 1) {
        float ov = __shfl_xor(v, off, 32);
        int   oi = __shfl_xor(idx, off, 32);
        if (ov > v || (ov == v && oi < idx)) { v = ov; idx = oi; }
    }
    __syncthreads();
    if ((threadIdx.x & 31) == 0) { sv[threadIdx.x >> 5] = v; si[threadIdx.x >> 5] = idx; }
    __syncthreads();
    if (threadIdx.x == 0) {
        float bestv = sv[0]; int besti = si[0];
        int nw = (blockDim.x + 31) >> 5;
        for (int w = 1; w < nw; ++w)
            if (sv[w] > bestv || (sv[w] == bestv && si[w] < besti)) { bestv = sv[w]; besti = si[w]; }
        bv = bestv; bi = besti;
    }
    __syncthreads();
    bv_out = bv; bi_out = bi;
    __syncthreads();
}

// ---------------- zero init ----------------
__global__ void zero_kernel(float* avg_probs, float* counts, float* sample_acc, float* vq_acc) {
    int t = blockIdx.x * blockDim.x + threadIdx.x;
    if (t < KC) { avg_probs[t] = 0.f; counts[t] = 0.f; }
    if (t == 0) { sample_acc[0] = 0.f; vq_acc[0] = 0.f; }
}

// ---------------- x (N,C,T) -> xf_bf16 (NT,C), LDS-tiled transpose ----------------
__global__ __launch_bounds__(256) void transpose_cast_kernel(
    const float* __restrict__ x, unsigned short* __restrict__ xf) {
    __shared__ float tile[32][33];
    const int n  = blockIdx.z;
    const int c0 = blockIdx.y * 32;
    const int t0 = blockIdx.x * 32;
    const int tx = threadIdx.x & 31;
    const int ty = threadIdx.x >> 5;   // 0..7
    #pragma unroll
    for (int s = 0; s < 32; s += 8) {
        int c = c0 + ty + s;
        tile[ty + s][tx] = x[((size_t)n * CC + c) * TT + t0 + tx];
    }
    __syncthreads();
    #pragma unroll
    for (int s = 0; s < 32; s += 8) {
        int t = t0 + ty + s;
        xf[((size_t)(n * TT + t)) * CC + c0 + tx] = f2bf(tile[tx][ty + s]);
    }
}

// ---------------- generic matrix transpose + cast: in (R x C) f32 -> out (C x R) bf16 ----
__global__ __launch_bounds__(256) void transpose_cast_mat_kernel(
    const float* __restrict__ in, unsigned short* __restrict__ out, int R, int C) {
    __shared__ float tile[32][33];
    const int r0 = blockIdx.y * 32;
    const int c0 = blockIdx.x * 32;
    const int tx = threadIdx.x & 31;
    const int ty = threadIdx.x >> 5;   // 0..7
    #pragma unroll
    for (int s = 0; s < 32; s += 8)
        tile[ty + s][tx] = in[(size_t)(r0 + ty + s) * C + c0 + tx];
    __syncthreads();
    #pragma unroll
    for (int s = 0; s < 32; s += 8)
        out[(size_t)(c0 + ty + s) * R + r0 + tx] = f2bf(tile[tx][ty + s]);
}

// ---------------- generic f32 -> bf16 cast ----------------
__global__ void cast_bf16_kernel(const float* __restrict__ in,
                                 unsigned short* __restrict__ out, int n) {
    int i = blockIdx.x * 256 + threadIdx.x;
    if (i < n) out[i] = f2bf(in[i]);
}

// ---------------- bf16 WMMA tiled GEMM: C = alpha * A @ B^T(mem) ----------------
// Block tile 64(M) x 128(N) x 64(K), double-buffered LDS, 8 waves,
// each wave computes a 16x64 strip => 4 accumulators, 8 WMMA per K-step.
// A: M x K row-major.  B memory: N x K row-major (k-contiguous rows).
// Tiles are staged global->LDS with GLOBAL_LOAD_ASYNC_TO_LDS_B128 (no VGPR
// staging, tracked by ASYNCcnt) when the builtin is available.
#define BM 64
#define BN 128
#define BK 64

__global__ __launch_bounds__(256) void gemm_bf16_wmma(
    const unsigned short* __restrict__ A,
    const unsigned short* __restrict__ B,   // N x K, k-contiguous
    float* __restrict__ C,
    int M, int N, int K, float alpha) {

    __shared__ unsigned short As[2][BM][BK];   // 16 KB
    __shared__ unsigned short Bs[2][BN][BK];   // 32 KB  (Bs[n][k], k-contiguous)

    const int tid  = threadIdx.x;
    const int lane = tid & 31;
    const int wave = tid >> 5;     // 0..7
    const int half = lane >> 4;    // 0/1
    const int l16  = lane & 15;

    const int m0 = blockIdx.y * BM;
    const int n0 = blockIdx.x * BN;
    const int waveM = wave & 3;    // 4 row tiles of 16
    const int waveN = wave >> 2;   // 2 col strips of 64

    v8f acc[4] = {v8f{}, v8f{}, v8f{}, v8f{}};

    const int nk = K >> 6;   // BK = 64

    // A: 512 chunks of 8 u16; chunk = tid*2+i -> row = chunk>>3, col = (chunk&7)*8
    // B: 1024 chunks;        chunk = tid*4+i -> n   = chunk>>3, k   = (chunk&7)*8
#if USE_ASYNC
    auto issueTile = [&](int kt, int buf) {
        const int k0 = kt << 6;
        #pragma unroll
        for (int i = 0; i < 2; ++i) {
            int chunk = tid * 2 + i;
            int r = chunk >> 3, c8 = (chunk & 7) << 3;
            async_copy_b128(A + (size_t)(m0 + r) * K + k0 + c8, &As[buf][r][c8]);
        }
        #pragma unroll
        for (int i = 0; i < 4; ++i) {
            int chunk = tid * 4 + i;
            int nn = chunk >> 3, k8 = (chunk & 7) << 3;
            async_copy_b128(B + (size_t)(n0 + nn) * K + k0 + k8, &Bs[buf][nn][k8]);
        }
    };
#else
    auto stageTile = [&](int kt, int buf) {
        const int k0 = kt << 6;
        #pragma unroll
        for (int i = 0; i < 2; ++i) {
            int chunk = tid * 2 + i;
            int r = chunk >> 3, c8 = (chunk & 7) << 3;
            uint4 v = *(const uint4*)(A + (size_t)(m0 + r) * K + k0 + c8);
            *(uint4*)&As[buf][r][c8] = v;   // short live range: no spill
        }
        #pragma unroll
        for (int i = 0; i < 4; ++i) {
            int chunk = tid * 4 + i;
            int nn = chunk >> 3, k8 = (chunk & 7) << 3;
            uint4 v = *(const uint4*)(B + (size_t)(n0 + nn) * K + k0 + k8);
            *(uint4*)&Bs[buf][nn][k8] = v;
        }
    };
#endif

#if USE_ASYNC
    issueTile(0, 0);
    wait_async();
    __syncthreads();
#else
    stageTile(0, 0);
    __syncthreads();
#endif

    const int mr = waveM * 16 + l16;

    for (int kt = 0; kt < nk; ++kt) {
        const int p = kt & 1;

#if USE_ASYNC
        if (kt + 1 < nk) issueTile(kt + 1, 1 - p);   // ASYNCcnt tracks; no VGPR staging
#else
        if (kt + 1 < nk) stageTile(kt + 1, 1 - p);   // barrier of iter kt-1 made 1-p safe
#endif

        // ---- compute from buffer p: 2 K-halves x 4 N-tiles = 8 WMMA ----
        #pragma unroll
        for (int kh = 0; kh < 2; ++kh) {
            const int kb = kh * 32;
            // A fragment: lane row M=l16; VGPR0-3 => K=8*half.., VGPR4-7 => K=16+8*half..
            v8bf alo = *(const v8bf*)&As[p][mr][kb + half * 8];
            v8bf ahi = *(const v8bf*)&As[p][mr][kb + 16 + half * 8];
            v16bf afrag = __builtin_shufflevector(alo, ahi,
                0, 1, 2, 3, 4, 5, 6, 7, 8, 9, 10, 11, 12, 13, 14, 15);
            #pragma unroll
            for (int j = 0; j < 4; ++j) {
                const int nr = waveN * 64 + j * 16 + l16;
                // B fragment: lane col N=l16; VGPR0-3 => K=8*half.., VGPR4-7 => K=16+8*half..
                v8bf blo = *(const v8bf*)&Bs[p][nr][kb + half * 8];
                v8bf bhi = *(const v8bf*)&Bs[p][nr][kb + 16 + half * 8];
                v16bf bfrag = __builtin_shufflevector(blo, bhi,
                    0, 1, 2, 3, 4, 5, 6, 7, 8, 9, 10, 11, 12, 13, 14, 15);
                acc[j] = __builtin_amdgcn_wmma_f32_16x16x32_bf16(
                    false, afrag, false, bfrag, (short)0, acc[j], false, false);
            }
        }

#if USE_ASYNC
        if (kt + 1 < nk) wait_async();   // next tile fully in LDS
#endif
        __syncthreads();
    }

    // ---- epilogue: C/D layout => VGPR v: M = v + 8*half, N = lane&15 ----
    const int rowBase = m0 + waveM * 16 + half * 8;
    #pragma unroll
    for (int v = 0; v < 8; ++v) {
        float* crow = C + (size_t)(rowBase + v) * N;
        #pragma unroll
        for (int j = 0; j < 4; ++j)
            crow[n0 + waveN * 64 + j * 16 + l16] = acc[j][v] * alpha;
    }
}

// ---------------- row L2 normalize + cast ----------------
__global__ __launch_bounds__(256) void rownorm_kernel(
    const float* __restrict__ in, unsigned short* __restrict__ out, int width) {
    const int r = blockIdx.x;
    const float* row = in + (size_t)r * width;
    float ss = 0.f;
    for (int j = threadIdx.x; j < width; j += 256) { float v = row[j]; ss += v * v; }
    float tot = block_reduce_sum(ss);
    float inv = 1.0f / (sqrtf(tot) + 1e-8f);
    unsigned short* orow = out + (size_t)r * width;
    for (int j = threadIdx.x; j < width; j += 256) orow[j] = f2bf(row[j] * inv);
}

// ---------------- softmax + argmax + entropy stats (K=1024, block=256) ----------------
#define RPB 8
__global__ __launch_bounds__(256) void softmax_stats_kernel(
    const float* __restrict__ logits,
    unsigned short* __restrict__ attn_bf,
    float* __restrict__ avg_probs,
    float* __restrict__ counts,
    float* __restrict__ sample_acc,
    float* __restrict__ code_out) {
    const int tid = threadIdx.x;
    float avg_loc[4] = {0.f, 0.f, 0.f, 0.f};
    float sent_loc = 0.f;

    for (int rr = 0; rr < RPB; ++rr) {
        const int r = blockIdx.x * RPB + rr;
        const float* row = logits + (size_t)r * KC;
        float l[4];
        float mx = -3.4e38f; int mi = 0;
        #pragma unroll
        for (int i = 0; i < 4; ++i) {
            int j = tid + 256 * i;
            l[i] = row[j];
            if (l[i] > mx) { mx = l[i]; mi = j; }
        }
        float bmax; int bidx;
        block_argmax(mx, mi, bmax, bidx);

        // softmax(logits) -> attention weights (bf16)
        float e[4]; float s = 0.f;
        #pragma unroll
        for (int i = 0; i < 4; ++i) { e[i] = __expf(l[i] - bmax); s += e[i]; }
        float stot = block_reduce_sum(s);
        float inv = 1.0f / stot;
        #pragma unroll
        for (int i = 0; i < 4; ++i)
            attn_bf[(size_t)r * KC + tid + 256 * i] = f2bf(e[i] * inv);

        // softmax(logits/TEMP) stats (1/TEMP = 100)
        float et[4]; float s2 = 0.f;
        #pragma unroll
        for (int i = 0; i < 4; ++i) { et[i] = __expf((l[i] - bmax) * 100.0f); s2 += et[i]; }
        float s2tot = block_reduce_sum(s2);
        float inv2 = 1.0f / s2tot;
        float lse = __logf(s2tot);
        #pragma unroll
        for (int i = 0; i < 4; ++i) {
            float p  = et[i] * inv2;
            float lp = (l[i] - bmax) * 100.0f - lse;
            sent_loc += p * lp;
            avg_loc[i] += p;
        }

        if (tid == 0) {
            atomicAdd(&counts[bidx], 1.0f);
            code_out[r] = (float)bidx;
        }
    }

    #pragma unroll
    for (int i = 0; i < 4; ++i) atomicAdd(&avg_probs[tid + 256 * i], avg_loc[i]);
    float st = block_reduce_sum(sent_loc);
    if (tid == 0) atomicAdd(sample_acc, st);
}

// ---------------- z_q -> x_d (transpose back) + vq_loss accumulation ----------------
__global__ __launch_bounds__(256) void zq_finalize_kernel(
    const float* __restrict__ zq, const float* __restrict__ x,
    float* __restrict__ xd_out, float* __restrict__ vq_acc) {
    size_t o = (size_t)blockIdx.x * 256 + threadIdx.x;
    int nt = (int)(o >> 9);      // / CC
    int c  = (int)(o & 511);
    int n  = nt >> 10;           // / TT
    int t  = nt & 1023;
    size_t xi = ((size_t)n * CC + c) * TT + t;
    float z  = zq[o];
    float xv = x[xi];
    xd_out[xi] = z;              // forward value of straight-through = z_q
    float d = z - xv;
    float s = block_reduce_sum(d * d);
    if (threadIdx.x == 0) atomicAdd(vq_acc, s);
}

// ---------------- final scalars ----------------
__global__ __launch_bounds__(256) void finalize_kernel(
    const float* __restrict__ avg_probs, const float* __restrict__ counts,
    const float* __restrict__ sample_acc, const float* __restrict__ vq_acc,
    float* __restrict__ out) {
    float asum = 0.f, psum = 0.f;
    #pragma unroll
    for (int i = 0; i < 4; ++i) {
        int j = threadIdx.x + 256 * i;
        float avg = avg_probs[j] * INV_NT;
        asum += avg * __logf(avg + 1e-5f);
        float pr = counts[j] * INV_NT;
        psum += pr * __logf(pr + 1e-7f);
    }
    float atot = block_reduce_sum(asum);
    float ptot = block_reduce_sum(psum);
    if (threadIdx.x == 0) {
        const size_t base = (size_t)NN * CC * TT;
        float sent = -sample_acc[0] * INV_NT;
        float aent = -atot;
        out[base + 0] = sent;               // sample_entropy
        out[base + 1] = aent;               // avg_entropy
        out[base + 2] = sent - aent;        // ent_loss
        out[base + 3] = __expf(-ptot);      // perplexity
        out[base + 4] = vq_acc[0] / (float)((size_t)NT * CC);  // vq_loss
    }
}

// ---------------- host launcher ----------------
extern "C" void kernel_launch(void* const* d_in, const int* in_sizes, int n_in,
                              void* d_out, int out_size, void* d_ws, size_t ws_size,
                              hipStream_t stream) {
    const float* x  = (const float*)d_in[0];
    const float* cb = (const float*)d_in[1];
    const float* Wq = (const float*)d_in[2];
    const float* Wk = (const float*)d_in[3];
    float* out = (float*)d_out;

    char* ws = (char*)d_ws;
    size_t off = 0;
    auto take = [&](size_t bytes) -> char* {
        char* p = ws + off;
        off = (off + bytes + 255) & ~(size_t)255;
        return p;
    };
    unsigned short* xf_bf   = (unsigned short*)take((size_t)NT * CC * 2);
    unsigned short* wqT_bf  = (unsigned short*)take((size_t)CC * AD * 2);  // AD x CC
    unsigned short* wkT_bf  = (unsigned short*)take((size_t)CC * AD * 2);  // AD x CC
    unsigned short* cb_bf   = (unsigned short*)take((size_t)KC * CC * 2);  // KC x CC (A of k-GEMM)
    unsigned short* cbT_bf  = (unsigned short*)take((size_t)KC * CC * 2);  // CC x KC (B of zq-GEMM)
    float*          q_f     = (float*)take((size_t)NT * AD * 4);
    float*          k_f     = (float*)take((size_t)KC * AD * 4);
    unsigned short* qn_bf   = (unsigned short*)take((size_t)NT * AD * 2);
    unsigned short* kn_bf   = (unsigned short*)take((size_t)KC * AD * 2);
    float*          logits  = (float*)take((size_t)NT * KC * 4);
    unsigned short* attn_bf = (unsigned short*)take((size_t)NT * KC * 2);
    float*          zq      = (float*)take((size_t)NT * CC * 4);
    float*          avg_pr  = (float*)take(KC * 4);
    float*          counts  = (float*)take(KC * 4);
    float*          s_acc   = (float*)take(4);
    float*          v_acc   = (float*)take(4);

    zero_kernel<<<4, 256, 0, stream>>>(avg_pr, counts, s_acc, v_acc);

    // casts / transposes (B matrices pre-transposed to N x K, k-contiguous)
    transpose_cast_kernel<<<dim3(TT / 32, CC / 32, NN), 256, 0, stream>>>(x, xf_bf);
    transpose_cast_mat_kernel<<<dim3(AD / 32, CC / 32), 256, 0, stream>>>(Wq, wqT_bf, CC, AD);
    transpose_cast_mat_kernel<<<dim3(AD / 32, CC / 32), 256, 0, stream>>>(Wk, wkT_bf, CC, AD);
    transpose_cast_mat_kernel<<<dim3(CC / 32, KC / 32), 256, 0, stream>>>(cb, cbT_bf, KC, CC);
    cast_bf16_kernel<<<(KC * CC + 255) / 256, 256, 0, stream>>>(cb, cb_bf, KC * CC);

    // q_pre = xf @ Wq ; k_pre = codebook @ Wk     (B given as N x K)
    gemm_bf16_wmma<<<dim3(AD / BN, NT / BM), 256, 0, stream>>>(xf_bf, wqT_bf, q_f, NT, AD, CC, 1.0f);
    gemm_bf16_wmma<<<dim3(AD / BN, KC / BM), 256, 0, stream>>>(cb_bf, wkT_bf, k_f, KC, AD, CC, 1.0f);

    rownorm_kernel<<<NT, 256, 0, stream>>>(q_f, qn_bf, AD);
    rownorm_kernel<<<KC, 256, 0, stream>>>(k_f, kn_bf, AD);

    // logits = (q @ kT) / sqrt(512)   (kn is KC x AD, already N x K layout)
    const float inv_sqrt_ad = 0.04419417382415922f;
    gemm_bf16_wmma<<<dim3(KC / BN, NT / BM), 256, 0, stream>>>(qn_bf, kn_bf, logits, NT, KC, AD, inv_sqrt_ad);

    float* code_out = out + (size_t)NN * CC * TT + 5;
    softmax_stats_kernel<<<NT / RPB, 256, 0, stream>>>(logits, attn_bf, avg_pr, counts, s_acc, code_out);

    // z_q = attn @ codebook   (cbT is CC x KC -> N x K layout)
    gemm_bf16_wmma<<<dim3(CC / BN, NT / BM), 256, 0, stream>>>(attn_bf, cbT_bf, zq, NT, CC, KC, 1.0f);

    zq_finalize_kernel<<<(NT * (size_t)CC) / 256, 256, 0, stream>>>(zq, x, out, v_acc);
    finalize_kernel<<<1, 256, 0, stream>>>(avg_pr, counts, s_acc, v_acc, out);
}
